// i2D_QKV_34806414967369
// MI455X (gfx1250) — matrix operand order repaired and checked by reading, not compiled, not run on hardware
//
#include <hip/hip_runtime.h>

// Problem constants from the reference
#define TT 4096
#define CC 512
#define LOG2GAMMA  (-0.04580368957f)   // log2(0.96875)
#define LOG2_10000 (13.287712379549449f)

// LDS strides (in 16-bit elements). 40 halfwords = 80B: 16B-aligned and
// rotates bank groups between rows (80B = 20 banks). 520 = 512 + 8: same.
#define XS_STRIDE 40
#define WS_STRIDE 40
#define VS_STRIDE 40
#define PB_STRIDE 40
#define QS_STRIDE 520

typedef __attribute__((ext_vector_type(16))) __bf16 v16bf;
typedef __attribute__((ext_vector_type(8)))  float  v8f;

static __device__ __forceinline__ unsigned short f2bf(float f) {
  unsigned int u = __float_as_uint(f);
  u += 0x7FFFu + ((u >> 16) & 1u);            // round-to-nearest-even
  return (unsigned short)(u >> 16);
}
static __device__ __forceinline__ unsigned int pack2bf(float a, float b) {
  return (unsigned int)f2bf(a) | ((unsigned int)f2bf(b) << 16);
}
static __device__ __forceinline__ float silu(float x) {
  return x / (1.0f + __expf(-x));
}

// Build a v16bf fragment from two 16B-aligned LDS chunks (2x ds_load_b128).
static __device__ __forceinline__ v16bf ld_frag2(const unsigned short* p0,
                                                 const unsigned short* p1) {
  union { uint4 q[2]; v16bf v; } u;
  u.q[0] = *(const uint4*)p0;
  u.q[1] = *(const uint4*)p1;
  return u.v;
}
// A fragment (16x32 bf16), ISA 7.12.2: lane(row M=lane%16) holds
// k = h*8 + [0,8) and k = 16 + h*8 + [0,8), h = lane/16. Both chunks contiguous.
static __device__ __forceinline__ v16bf load_a_frag(const unsigned short* row, int h) {
  return ld_frag2(row + h * 8, row + 16 + h * 8);
}
// B fragment (32x16 bf16): lane(col N=lane%16) holds k = h*8 + [0,8) and k = 16 + h*8 + [0,8).
// `colbase` must point at this column's K-contiguous 32-element run.
static __device__ __forceinline__ v16bf load_b_frag(const unsigned short* colbase, int h) {
  return ld_frag2(colbase + h * 8, colbase + 16 + h * 8);
}

// ---------------------------------------------------------------------------
// Kernel 1: Q/K/V = epilogue(X @ W).  64x64 tile per block, 8 waves, bf16 WMMA.
// W tile stored K-major (transposed) in LDS so B fragments are 2x b128 loads.
// RoPE on channels [0,64) of Q,K (== col-tile 0), then SiLU.
// V is written TRANSPOSED to global: V^T[b][c][t], so kernel 2 gets K-major V.
// ---------------------------------------------------------------------------
__global__ __launch_bounds__(256) void qkv_proj_kernel(
    const float* __restrict__ X,  const float* __restrict__ Wq,
    const float* __restrict__ Wk, const float* __restrict__ Wv,
    unsigned short* __restrict__ Qb, unsigned short* __restrict__ Kb,
    unsigned short* __restrict__ Vt) {
  __shared__ unsigned short Xs[64 * XS_STRIDE];   // 64 rows x 32 k (padded)
  __shared__ unsigned short WsT[64 * WS_STRIDE];  // 64 cols x 32 k (K-major)
  __shared__ float          Eb[64][64];           // f32 accum staging

  const int tid  = threadIdx.x;
  const int w    = tid >> 5, lane = tid & 31;
  const int h    = lane >> 4, mn  = lane & 15;
  const int rowBase = blockIdx.x * 64;
  const int colBase = blockIdx.y * 64;
  const int wid     = blockIdx.z;
  const float* W = (wid == 0) ? Wq : ((wid == 1) ? Wk : Wv);

  const int tIdx = w * 2;                  // 16 tiles (4x4), 2 per wave
  const int tm = tIdx >> 2, tn0 = tIdx & 3, tn1 = tn0 + 1;

  v8f acc0 = {}, acc1 = {};

  for (int kb = 0; kb < CC; kb += 32) {
    { // X tile: one thread packs 8 contiguous floats -> one ds_store_b128
      const int r  = tid >> 2;
      const int c8 = (tid & 3) * 8;
      const float* xs = X + (size_t)(rowBase + r) * CC + kb + c8;
      uint4 pk;
      pk.x = pack2bf(xs[0], xs[1]); pk.y = pack2bf(xs[2], xs[3]);
      pk.z = pack2bf(xs[4], xs[5]); pk.w = pack2bf(xs[6], xs[7]);
      *((uint4*)(Xs + r * XS_STRIDE) + (tid & 3)) = pk;
    }
    { // W tile: coalesced global read along n, transposed scatter into LDS
      const int k  = tid >> 3;
      const int n0 = (tid & 7) * 8;
      const float* ws = W + (size_t)(kb + k) * CC + colBase + n0;
#pragma unroll
      for (int j = 0; j < 8; ++j)
        WsT[(n0 + j) * WS_STRIDE + k] = f2bf(ws[j]);
    }
    __syncthreads();
    v16bf a  = load_a_frag(Xs + (tm * 16 + mn) * XS_STRIDE, h);
    v16bf b0 = load_b_frag(WsT + (tn0 * 16 + mn) * WS_STRIDE, h);
    v16bf b1 = load_b_frag(WsT + (tn1 * 16 + mn) * WS_STRIDE, h);
    acc0 = __builtin_amdgcn_wmma_f32_16x16x32_bf16(false, a, false, b0, (short)0, acc0, false, false);
    acc1 = __builtin_amdgcn_wmma_f32_16x16x32_bf16(false, a, false, b1, (short)0, acc1, false, false);
    __syncthreads();
  }

  // C/D layout: element v -> row = v + 8*h, col = lane%16
#pragma unroll
  for (int v = 0; v < 8; ++v) {
    Eb[tm * 16 + h * 8 + v][tn0 * 16 + mn] = acc0[v];
    Eb[tm * 16 + h * 8 + v][tn1 * 16 + mn] = acc1[v];
  }
  __syncthreads();

  const int r    = tid >> 2;
  const int cseg = (tid & 3) * 16;
  const int gr   = rowBase + r;
  const int pos  = gr & (TT - 1);          // position restarts per batch
  if (wid == 2) {
    // V: SiLU then store transposed: V^T[(b*512 + c) * 4096 + t]
    const size_t bc = (size_t)(gr >> 12) * CC;   // batch * 512
    const int    t  = gr & (TT - 1);
#pragma unroll
    for (int p = 0; p < 16; ++p) {
      int gc = colBase + cseg + p;
      Vt[(bc + gc) * TT + t] = f2bf(silu(Eb[r][cseg + p]));
    }
  } else if (colBase == 0) {
    // Q/K col-tile 0 == the RoPE region (ROPE_DIM = 64)
    unsigned short* dst = (wid == 0) ? Qb : Kb;
#pragma unroll
    for (int p = 0; p < 8; ++p) {
      int c0 = cseg + 2 * p;               // global channel (colBase==0)
      float invf = exp2f(-((float)c0 / 64.0f) * LOG2_10000);
      float fr = (float)pos * invf;
      float sn, cs;
      __sincosf(fr, &sn, &cs);
      float x1 = Eb[r][c0], x2 = Eb[r][c0 + 1];
      dst[(size_t)gr * CC + c0]     = f2bf(silu(x1 * cs - x2 * sn));
      dst[(size_t)gr * CC + c0 + 1] = f2bf(silu(x2 * cs + x1 * sn));
    }
  } else {
    unsigned short* dst = (wid == 0) ? Qb : Kb;
#pragma unroll
    for (int p = 0; p < 16; ++p)
      dst[(size_t)gr * CC + colBase + cseg + p] = f2bf(silu(Eb[r][cseg + p]));
  }
}

// ---------------------------------------------------------------------------
// Kernel 2: causal retention + fused GroupNorm.
// Block = 32-row tile of one batch; loop s-tiles 0..rt.
//   P = Q K^T (k split across 8 waves, ds_add_f32 reduction in LDS)
//   P *= gamma^(t-s) masked, requantize bf16
//   out += P V (V already K-major in LDS via transposed global layout)
// Epilogue: fused GroupNorm (8 groups x 64 ch), fp32 out.
// ---------------------------------------------------------------------------
__global__ __launch_bounds__(256) void retention_gn_kernel(
    const unsigned short* __restrict__ Qb, const unsigned short* __restrict__ Kb,
    const unsigned short* __restrict__ Vt, const float* __restrict__ gnw,
    const float* __restrict__ gnb, float* __restrict__ out) {
  extern __shared__ unsigned char smem[];
  unsigned short* Qs  = (unsigned short*)(smem);           // 32 x 520  (33280 B)
  unsigned short* Ks  = (unsigned short*)(smem + 33280);   // 32 x 520  (33280 B)
  unsigned short* VsT = (unsigned short*)(smem + 66560);   // 512 x 40  (40960 B)
  float*          Pf  = (float*)(smem + 107520);           // 32 x 32   (4096 B)
  unsigned short* Pb  = (unsigned short*)(smem + 111616);  // 32 x 40   (2560 B)

  const int tid = threadIdx.x;
  const int w   = tid >> 5, lane = tid & 31;
  const int h   = lane >> 4, mn  = lane & 15;
  const int rt  = blockIdx.x;                 // row tile within batch
  const int b   = blockIdx.y;
  const size_t rowG = (size_t)b * TT + (size_t)rt * 32;

  { // pin Q tile (32x512 bf16) in LDS: b128 copies, padded rows
#pragma unroll
    for (int e = 0; e < 8; ++e) {
      int lin = tid * 8 + e;
      int r = lin >> 6, j = lin & 63;
      *((uint4*)(Qs + r * QS_STRIDE) + j) =
          *((const uint4*)(Qb + (rowG + r) * CC) + j);
    }
  }

  v8f zero = {};
  v8f o[8];
#pragma unroll
  for (int j = 0; j < 8; ++j) o[j] = zero;
  const int tm2 = (w >= 4) ? 1 : 0;           // out tileM owned by this wave

  for (int st = 0; st <= rt; ++st) {
    const size_t srowG = (size_t)b * TT + (size_t)st * 32;
    { // K tile (row-major) and V tile (already K-major in global V^T)
#pragma unroll
      for (int e = 0; e < 8; ++e) {
        int lin = tid * 8 + e;
        int r = lin >> 6, j = lin & 63;
        *((uint4*)(Ks + r * QS_STRIDE) + j) =
            *((const uint4*)(Kb + (srowG + r) * CC) + j);
      }
#pragma unroll
      for (int e = 0; e < 8; ++e) {
        int lin = tid * 8 + e;                 // 2048 uint4 = 512 rows x 4
        int c = lin >> 2, j = lin & 3;
        *((uint4*)(VsT + c * VS_STRIDE) + j) =
            *((const uint4*)(Vt + ((size_t)b * CC + c) * TT + st * 32) + j);
      }
#pragma unroll
      for (int e = 0; e < 4; ++e) Pf[tid * 4 + e] = 0.0f;
      if (st < rt)  // hint next K tile toward cache
        __builtin_prefetch(Kb + (srowG + 32) * CC, 0, 1);
    }
    __syncthreads();

    // ---- P partial: this wave covers k in [w*64, w*64+64) ----
    v8f pacc[4];
#pragma unroll
    for (int q = 0; q < 4; ++q) pacc[q] = zero;
#pragma unroll
    for (int cc = 0; cc < 2; ++cc) {
      const int kb = w * 64 + cc * 32;
#pragma unroll
      for (int pm = 0; pm < 2; ++pm) {
        v16bf a = load_a_frag(Qs + (pm * 16 + mn) * QS_STRIDE + kb, h);
#pragma unroll
        for (int pn = 0; pn < 2; ++pn) {
          v16bf bb = load_b_frag(Ks + (pn * 16 + mn) * QS_STRIDE + kb, h);
          pacc[pm * 2 + pn] = __builtin_amdgcn_wmma_f32_16x16x32_bf16(
              false, a, false, bb, (short)0, pacc[pm * 2 + pn], false, false);
        }
      }
    }
    // cross-wave reduction of P via LDS float atomics (ds_add_f32)
#pragma unroll
    for (int pm = 0; pm < 2; ++pm)
#pragma unroll
      for (int pn = 0; pn < 2; ++pn)
#pragma unroll
        for (int v = 0; v < 8; ++v)
          atomicAdd(&Pf[(pm * 16 + h * 8 + v) * 32 + pn * 16 + mn],
                    pacc[pm * 2 + pn][v]);
    __syncthreads();

    // ---- decay mask + requantize P to bf16 ----
#pragma unroll
    for (int e = 0; e < 4; ++e) {
      int lin = tid * 4 + e;
      int rr = lin >> 5, cc2 = lin & 31;
      int d = (rt * 32 + rr) - (st * 32 + cc2);
      float pv = (d < 0) ? 0.0f : Pf[lin] * exp2f((float)d * LOG2GAMMA);
      Pb[rr * PB_STRIDE + cc2] = f2bf(pv);
    }
    __syncthreads();

    // ---- out += P @ V : 8 out tiles per wave, K=32 -> one WMMA each ----
    v16bf a2 = load_a_frag(Pb + (tm2 * 16 + mn) * PB_STRIDE, h);
#pragma unroll
    for (int j = 0; j < 8; ++j) {
      int tn2 = (w * 8 + j) & 31;
      v16bf bb = load_b_frag(VsT + (tn2 * 16 + mn) * VS_STRIDE, h);
      o[j] = __builtin_amdgcn_wmma_f32_16x16x32_bf16(false, a2, false, bb,
                                                     (short)0, o[j], false, false);
    }
    __syncthreads();   // protect Ks/VsT/Pf before next iteration / epilogue
  }

  // ---- stage result (reuse Ks+VsT region: 32x512 f32 = 64KB) ----
  float* ob = (float*)(smem + 33280);
#pragma unroll
  for (int j = 0; j < 8; ++j) {
    int tn2 = (w * 8 + j) & 31;
#pragma unroll
    for (int v = 0; v < 8; ++v)
      ob[(tm2 * 16 + h * 8 + v) * CC + tn2 * 16 + mn] = o[j][v];
  }
  __syncthreads();

  // ---- fused GroupNorm: one thread per (row, group) of 64 channels ----
  {
    const int rr = tid >> 3;
    const int g  = tid & 7;
    const int cb = g * 64;
    float s = 0.0f, s2 = 0.0f;
#pragma unroll 8
    for (int c = 0; c < 64; ++c) {
      float x = ob[rr * CC + cb + c];
      s += x; s2 += x * x;
    }
    float mu  = s * (1.0f / 64.0f);
    float var = s2 * (1.0f / 64.0f) - mu * mu;
    float rs  = rsqrtf(var + 1e-5f);
    size_t orow = (rowG + rr) * CC;
#pragma unroll 8
    for (int c = 0; c < 64; ++c) {
      float x = (ob[rr * CC + cb + c] - mu) * rs;
      out[orow + cb + c] = x * gnw[cb + c] + gnb[cb + c];
    }
  }
}

// ---------------------------------------------------------------------------
extern "C" void kernel_launch(void* const* d_in, const int* in_sizes, int n_in,
                              void* d_out, int out_size, void* d_ws, size_t ws_size,
                              hipStream_t stream) {
  const float* X  = (const float*)d_in[0];
  const float* Wq = (const float*)d_in[1];
  const float* Wk = (const float*)d_in[2];
  const float* Wv = (const float*)d_in[3];
  const float* gw = (const float*)d_in[4];
  const float* gb = (const float*)d_in[5];
  float* out = (float*)d_out;

  // Workspace: Q,K row-major bf16; V transposed bf16 (each 8192x512 = 8MB)
  unsigned short* Qb = (unsigned short*)d_ws;
  unsigned short* Kb = Qb + (size_t)8192 * 512;
  unsigned short* Vt = Kb + (size_t)8192 * 512;

  // QKV projections + RoPE + SiLU -> bf16 (V stored transposed)
  qkv_proj_kernel<<<dim3(128, 8, 3), 256, 0, stream>>>(X, Wq, Wk, Wv, Qb, Kb, Vt);

  // Causal retention + decay + fused GroupNorm (dynamic LDS = 114176 B)
  retention_gn_kernel<<<dim3(128, 2), 256, 114176, stream>>>(Qb, Kb, Vt, gw, gb, out);
}
